// SceneDynamicContext_27032524161356
// MI455X (gfx1250) — matrix operand order repaired and checked by reading, not compiled, not run on hardware
//
#include <hip/hip_runtime.h>
#include <hip/hip_bf16.h>
#include <math.h>

// ---------------------------------------------------------------------------
// SceneDynamicContext forward for MI455X (gfx1250).
// Dense linear algebra -> v_wmma_f32_16x16x32_bf16 (fp32 tiles converted to
// bf16 in LDS, fp32 accumulation). Block tile 128x128, 8 waves x (32x64),
// double-buffered LDS with software-pipelined staging:
//   global loads (kt+1) -> WMMA (kt) -> LDS stores (kt+1) -> barrier
// ---------------------------------------------------------------------------

typedef __bf16 bf16_t;
typedef __attribute__((ext_vector_type(16))) __bf16 v16bf;
typedef __attribute__((ext_vector_type(8)))  float  v8f;

constexpr int Nn = 2048, Ee = 32768, Hh = 256, Dd = 2048, SDd = 100, Cc = 151, Rr = 51, ITERS = 3;

constexpr int BM = 128, BN = 128, BK = 32, TPB = 256;
constexpr int LDSK = 36; // 72B row stride: 8B-aligned vector ds ops, benign bank pattern

enum { EPI_NONE = 0, EPI_RELU = 1, EPI_MULTANH = 2 };

union FragBF { v16bf v; unsigned int w[8]; };

// C[M x Ncol] = epi( A[M x K] @ op(B) + bias * rowScale )
//   bIsWT=1: B is [Ncol x K] row-major (weights, op=B^T)
//   bIsWT=0: B is [K x Ncol] row-major
__global__ __launch_bounds__(TPB) void gemm_wmma_kernel(
    const float* __restrict__ A, int lda,
    const float* __restrict__ B, int ldb, int bIsWT,
    const float* __restrict__ bias,
    const float* __restrict__ rowScale,
    const float* __restrict__ aux,
    float* __restrict__ C, int ldc,
    int M, int Ncol, int K, int epi, int accum)
{
  __shared__ __align__(16) bf16_t As[2][BM][LDSK];
  __shared__ __align__(16) bf16_t Bs[2][BN][LDSK];
  const int t    = threadIdx.x;
  const int bm   = blockIdx.y * BM;
  const int bn   = blockIdx.x * BN;
  const int wave = t >> 5;
  const int lane = t & 31;
  const int wm   = wave & 3;   // 4 waves along M (32 rows each)
  const int wn   = wave >> 2;  // 2 waves along N (64 cols each)
  const int half = lane >> 4;
  const int l16  = lane & 15;

  const bool fullM = (bm + BM <= M);
  const bool fullN = (bn + BN <= Ncol);
  const bool aVec  = fullM && ((lda & 3) == 0);
  const bool bVec  = fullN && ((ldb & 3) == 0);

  const int nk = (K + BK - 1) / BK;
  const int rA = t >> 3;            // 0..31
  const int cA = (t & 7) * 4;       // 0,4,..,28

  // ---- staging: global -> regs ------------------------------------------
  auto loadA = [&](int kt, float (&ra)[16]) {
    int k0 = kt * BK;
    if (aVec && (k0 + BK <= K)) {
      #pragma unroll
      for (int i = 0; i < 4; ++i) {
        float4 v = *(const float4*)(A + (size_t)(bm + rA + 32 * i) * lda + k0 + cA);
        ra[4*i+0] = v.x; ra[4*i+1] = v.y; ra[4*i+2] = v.z; ra[4*i+3] = v.w;
      }
    } else {
      #pragma unroll
      for (int i = 0; i < 4; ++i) {
        int gr = bm + rA + 32 * i;
        int grc = gr < M ? gr : M - 1;
        #pragma unroll
        for (int j = 0; j < 4; ++j) {
          int gc = k0 + cA + j;
          int gcc = gc < K ? gc : K - 1;
          float v = A[(size_t)grc * lda + gcc];
          ra[4*i+j] = (gr < M && gc < K) ? v : 0.0f;
        }
      }
    }
  };
  auto loadB = [&](int kt, float (&rb)[16]) {
    int k0 = kt * BK;
    if (bIsWT) {
      if (bVec && (k0 + BK <= K)) {
        #pragma unroll
        for (int i = 0; i < 4; ++i) {
          float4 v = *(const float4*)(B + (size_t)(bn + rA + 32 * i) * ldb + k0 + cA);
          rb[4*i+0] = v.x; rb[4*i+1] = v.y; rb[4*i+2] = v.z; rb[4*i+3] = v.w;
        }
      } else {
        #pragma unroll
        for (int i = 0; i < 4; ++i) {
          int gn = bn + rA + 32 * i;
          int gnc = gn < Ncol ? gn : Ncol - 1;
          #pragma unroll
          for (int j = 0; j < 4; ++j) {
            int gk = k0 + cA + j;
            int gkc = gk < K ? gk : K - 1;
            float v = B[(size_t)gnc * ldb + gkc];
            rb[4*i+j] = (gn < Ncol && gk < K) ? v : 0.0f;
          }
        }
      }
    } else { // B row-major [K x Ncol]: rb[i] holds B[k0 + 2*i + t/128][bn + t%128], coalesced along n
      if (fullN && (k0 + BK <= K)) {
        #pragma unroll
        for (int i = 0; i < 16; ++i) {
          int idx = t + TPB * i;
          int n = idx & 127, c = idx >> 7;
          rb[i] = B[(size_t)(k0 + c) * ldb + bn + n];
        }
      } else {
        #pragma unroll
        for (int i = 0; i < 16; ++i) {
          int idx = t + TPB * i;
          int n = idx & 127, c = idx >> 7;
          int gn = bn + n, gk = k0 + c;
          int gnc = gn < Ncol ? gn : Ncol - 1;
          int gkc = gk < K ? gk : K - 1;
          float v = B[(size_t)gkc * ldb + gnc];
          rb[i] = (gn < Ncol && gk < K) ? v : 0.0f;
        }
      }
    }
  };
  // ---- staging: regs -> LDS (bf16) --------------------------------------
  auto storeA = [&](const float (&ra)[16], int buf) {
    #pragma unroll
    for (int i = 0; i < 4; ++i) {
      bf16_t* d = &As[buf][rA + 32 * i][cA];
      d[0] = (bf16_t)ra[4*i+0]; d[1] = (bf16_t)ra[4*i+1];
      d[2] = (bf16_t)ra[4*i+2]; d[3] = (bf16_t)ra[4*i+3];
    }
  };
  auto storeB = [&](const float (&rb)[16], int buf) {
    if (bIsWT) {
      #pragma unroll
      for (int i = 0; i < 4; ++i) {
        bf16_t* d = &Bs[buf][rA + 32 * i][cA];
        d[0] = (bf16_t)rb[4*i+0]; d[1] = (bf16_t)rb[4*i+1];
        d[2] = (bf16_t)rb[4*i+2]; d[3] = (bf16_t)rb[4*i+3];
      }
    } else {
      #pragma unroll
      for (int i = 0; i < 16; ++i) {
        int idx = t + TPB * i;
        int n = idx & 127, c = idx >> 7;
        Bs[buf][n][c] = (bf16_t)rb[i];
      }
    }
  };

  v8f acc[2][4] = {};
  float ra[16], rb[16];

  loadA(0, ra); loadB(0, rb);
  storeA(ra, 0); storeB(rb, 0);
  __syncthreads();

  for (int kt = 0; kt < nk; ++kt) {
    const int cur = kt & 1;
    const bool hasNext = (kt + 1 < nk);
    if (hasNext) { loadA(kt + 1, ra); loadB(kt + 1, rb); }  // loads in flight during WMMA

    // --- fragments per CDNA5 16-bit WMMA lane layouts
    FragBF a0, a1, bfr[4];
    const int ar0 = wm * 32 + l16;
    const int ar1 = ar0 + 16;
    #pragma unroll
    for (int v = 0; v < 8; ++v) {
      // A (16x32): lanes 0-15 hold K {0..7,16..23}; lanes 16-31 hold K {8..15,24..31}
      int kbA = ((v < 4) ? 0 : 16) + 8 * half + 2 * (v & 3);
      a0.w[v] = *(const unsigned int*)&As[cur][ar0][kbA];
      a1.w[v] = *(const unsigned int*)&As[cur][ar1][kbA];
      // B (32x16): lanes 0-15 hold K {0..7,16..23}; lanes 16-31 hold K {8..15,24..31}
      int kbB = ((v < 4) ? 0 : 16) + 8 * half + 2 * (v & 3);
      #pragma unroll
      for (int j = 0; j < 4; ++j)
        bfr[j].w[v] = *(const unsigned int*)&Bs[cur][wn * 64 + j * 16 + l16][kbB];
    }
    #pragma unroll
    for (int j = 0; j < 4; ++j) {
      acc[0][j] = __builtin_amdgcn_wmma_f32_16x16x32_bf16(false, a0.v, false, bfr[j].v, (short)0, acc[0][j], false, false);
      acc[1][j] = __builtin_amdgcn_wmma_f32_16x16x32_bf16(false, a1.v, false, bfr[j].v, (short)0, acc[1][j], false, false);
    }

    if (hasNext) { storeA(ra, cur ^ 1); storeB(rb, cur ^ 1); }  // wait lands here
    __syncthreads();
  }

  // --- epilogue: lane -> col = l16 (+16j), VGPR r -> row = r + 8*half
  #pragma unroll
  for (int i = 0; i < 2; ++i) {
    #pragma unroll
    for (int j = 0; j < 4; ++j) {
      int col = bn + wn * 64 + j * 16 + l16;
      if (col >= Ncol) continue;
      float bcol = bias ? bias[col] : 0.0f;
      #pragma unroll
      for (int r = 0; r < 8; ++r) {
        int row = bm + wm * 32 + i * 16 + half * 8 + r;
        if (row >= M) continue;
        float x = acc[i][j][r];
        x += bcol * (rowScale ? rowScale[row] : 1.0f);
        if (epi == EPI_RELU)         x = x > 0.0f ? x : 0.0f;
        else if (epi == EPI_MULTANH) x = tanhf(x * aux[(size_t)row * ldc + col]);
        size_t o = (size_t)row * ldc + col;
        if (accum) C[o] += x; else C[o] = x;
      }
    }
  }
}

// ----------------------------- helper kernels ------------------------------

__global__ void fill_kernel(float* p, float v, long long n) {
  long long i = (long long)blockIdx.x * blockDim.x + threadIdx.x;
  long long s = (long long)gridDim.x * blockDim.x;
  for (; i < n; i += s) p[i] = v;
}

__global__ void softmax_kernel(const float* __restrict__ in, float* __restrict__ out, int cols) {
  __shared__ float red[TPB];
  int row = blockIdx.x;
  const float* ir = in + (size_t)row * cols;
  float* orow = out + (size_t)row * cols;
  float m = -INFINITY;
  for (int c = threadIdx.x; c < cols; c += blockDim.x) m = fmaxf(m, ir[c]);
  red[threadIdx.x] = m; __syncthreads();
  for (int s = TPB / 2; s > 0; s >>= 1) { if (threadIdx.x < s) red[threadIdx.x] = fmaxf(red[threadIdx.x], red[threadIdx.x + s]); __syncthreads(); }
  m = red[0]; __syncthreads();
  float sum = 0.0f;
  for (int c = threadIdx.x; c < cols; c += blockDim.x) { float e = expf(ir[c] - m); orow[c] = e; sum += e; }
  red[threadIdx.x] = sum; __syncthreads();
  for (int s = TPB / 2; s > 0; s >>= 1) { if (threadIdx.x < s) red[threadIdx.x] += red[threadIdx.x + s]; __syncthreads(); }
  float inv = 1.0f / red[0];
  for (int c = threadIdx.x; c < cols; c += blockDim.x) orow[c] *= inv;
}

// out[e] = [ A[rowA][aOff : aOff+HA] | B[rowB][bOff : bOff+HB] ], rowX = idx or e
__global__ void concat_gather_kernel(float* __restrict__ out,
    const float* __restrict__ A, const int* __restrict__ aIdx, int aIdxStride, int aLd, int aOff, int HA,
    const float* __restrict__ B, const int* __restrict__ bIdx, int bIdxStride, int bLd, int bOff, int HB)
{
  int e = blockIdx.x;
  int ra = aIdx ? aIdx[(size_t)e * aIdxStride] : e;
  int rb = bIdx ? bIdx[(size_t)e * bIdxStride] : e;
  int W = HA + HB;
  for (int c = threadIdx.x; c < W; c += blockDim.x) {
    float v = (c < HA) ? A[(size_t)ra * aLd + aOff + c]
                       : B[(size_t)rb * bLd + bOff + (c - HA)];
    out[(size_t)e * W + c] = v;
  }
}

// u[e] = hid[e,:] . w + b ; optional exp + segment-denominator accumulation
__global__ void edge_dot_kernel(const float* __restrict__ hid, int ld,
    const float* __restrict__ w, const float* __restrict__ bptr,
    float* __restrict__ u, float* __restrict__ uexp,
    const int* __restrict__ idx, int idxStride, float* __restrict__ denom,
    int E, int K)
{
  int lane = threadIdx.x & 31;
  int e = blockIdx.x * (blockDim.x >> 5) + (threadIdx.x >> 5);
  if (e >= E) return;
  float s = 0.0f;
  for (int k = lane; k < K; k += 32) s += hid[(size_t)e * ld + k] * w[k];
  for (int off = 16; off > 0; off >>= 1) s += __shfl_down(s, off, 32);
  if (lane == 0) {
    float uu = s + bptr[0];
    if (u) u[e] = uu;
    if (uexp) {
      float ue = expf(uu);
      uexp[e] = ue;
      atomicAdd(&denom[idx[(size_t)e * idxStride]], ue);
    }
  }
}

__global__ void scatter_oo_kernel(const float* __restrict__ u, const int* __restrict__ rel,
                                  float* __restrict__ oo, int E, int Nv) {
  int e = blockIdx.x * blockDim.x + threadIdx.x;
  if (e < E) {
    int s = rel[2 * e], o = rel[2 * e + 1];
    oo[(size_t)s * Nv + o] = u[e];  // .set semantics: duplicate pairs -> one wins
  }
}

// per row: e = exp(v)*(v!=0); norm = e/(sum+1e-8) (+1 on diag); rs = sum/(sum+1e-8)+1
__global__ void oo_norm_kernel(float* __restrict__ oo, float* __restrict__ rs, int Nv) {
  __shared__ float red[TPB];
  int s = blockIdx.x;
  float* row = oo + (size_t)s * Nv;
  float sum = 0.0f;
  for (int c = threadIdx.x; c < Nv; c += blockDim.x) {
    float v = row[c];
    float e = (v != 0.0f) ? expf(v) : 0.0f;
    row[c] = e; sum += e;
  }
  red[threadIdx.x] = sum; __syncthreads();
  for (int t = TPB / 2; t > 0; t >>= 1) { if (threadIdx.x < t) red[threadIdx.x] += red[threadIdx.x + t]; __syncthreads(); }
  float total = red[0];
  float inv = 1.0f / (total + 1e-8f);
  for (int c = threadIdx.x; c < Nv; c += blockDim.x) {
    float nv = row[c] * inv;
    if (c == s) nv += 1.0f;   // + eye
    row[c] = nv;
  }
  if (threadIdx.x == 0) rs[s] = total * inv + 1.0f; // rowsum(norm_oo) incl. eye
}

// alpha = uexp/(denom[idx]+1e-8); agg[idx] += alpha*pf[e]; ssum[idx] += alpha
__global__ void alpha_agg_kernel(const float* __restrict__ uexp,
    const int* __restrict__ idx, int idxStride, const float* __restrict__ denom,
    const float* __restrict__ pf, float* __restrict__ agg, float* __restrict__ ssum, int Hd)
{
  int e = blockIdx.x;
  int n = idx[(size_t)e * idxStride];
  float a = uexp[e] / (denom[n] + 1e-8f);
  if (threadIdx.x == 0) atomicAdd(&ssum[n], a);
  for (int h = threadIdx.x; h < Hd; h += blockDim.x)
    atomicAdd(&agg[(size_t)n * Hd + h], a * pf[(size_t)e * Hd + h]);
}

__global__ void build_x_kernel(const float* __restrict__ fm, const float* __restrict__ msg,
    const float* __restrict__ se, float* __restrict__ x, int Dv, int SDv)
{
  int n = blockIdx.x;
  int W = Dv + SDv;
  for (int c = threadIdx.x; c < W; c += blockDim.x) {
    float v = (c < Dv) ? (fm[(size_t)n * Dv + c] + msg[(size_t)n * Dv + c])
                       : se[(size_t)n * SDv + (c - Dv)];
    x[(size_t)n * W + c] = v;
  }
}

__device__ __forceinline__ float sigm(float x) { return 1.0f / (1.0f + expf(-x)); }

__global__ void lstm_kernel(const float* __restrict__ pi, const float* __restrict__ ps,
                            float* __restrict__ memory, float* __restrict__ state, int Nv, int Hd)
{
  int i = blockIdx.x * blockDim.x + threadIdx.x;
  if (i >= Nv * Hd) return;
  int n = i / Hd, h = i - n * Hd;
  const float* pr = pi + (size_t)n * 6 * Hd;
  const float* sr = ps + (size_t)n * 5 * Hd;
  float ig = sigm(pr[h]          + sr[h]);
  float fg = sigm(pr[Hd + h]     + sr[Hd + h]);
  float mi = tanhf(pr[2 * Hd + h] + sr[2 * Hd + h]);
  float og = sigm(pr[3 * Hd + h] + sr[3 * Hd + h]);
  float mem = ig * mi + fg * memory[i];
  memory[i] = mem;
  float out = og * tanhf(mem);
  float hg = sigm(pr[4 * Hd + h] + sr[4 * Hd + h]);
  state[i] = hg * out + (1.0f - hg) * pr[5 * Hd + h];
}

__global__ void pf_update_kernel(float* __restrict__ pf, const float* __restrict__ srs,
                                 const float* __restrict__ sro, const int* __restrict__ rel, int Hd)
{
  int e = blockIdx.x;
  int s = rel[2 * e], o = rel[2 * e + 1];
  for (int h = threadIdx.x; h < Hd; h += blockDim.x)
    pf[(size_t)e * Hd + h] += srs[(size_t)s * Hd + h] + sro[(size_t)o * Hd + h];
}

// ------------------------------- host side ---------------------------------

static inline void gemm(hipStream_t s, const float* A, int lda, const float* B, int ldb, int bIsWT,
                        const float* bias, const float* rowScale, const float* aux,
                        float* C, int ldc, int M, int Ncol, int K, int epi, int accum)
{
  dim3 g((Ncol + BN - 1) / BN, (M + BM - 1) / BM);
  gemm_wmma_kernel<<<g, dim3(TPB), 0, s>>>(A, lda, B, ldb, bIsWT, bias, rowScale, aux,
                                           C, ldc, M, Ncol, K, epi, accum);
}

extern "C" void kernel_launch(void* const* d_in, const int* in_sizes, int n_in,
                              void* d_out, int out_size, void* d_ws, size_t ws_size,
                              hipStream_t stream) {
  (void)in_sizes; (void)n_in; (void)out_size; (void)ws_size;
  // Input order: jax tree-leaf order (dict keys sorted alphabetically, nested; 'b' < 'w')
  const float* last_rel  = (const float*)d_in[0];   // last_rel_input  E x 3H
  const float* obj_fmaps = (const float*)d_in[1];   // N x D
  const float* obj_logit = (const float*)d_in[2];   // N x C
  const float* Wo_b  = (const float*)d_in[3];  const float* Wo_w  = (const float*)d_in[4];   // W_o  (D,H)
  const float* Wor_b = (const float*)d_in[5];  const float* Wor_w = (const float*)d_in[6];   // W_or (D,H)
  const float* Wro_b = (const float*)d_in[7];  const float* Wro_w = (const float*)d_in[8];   // W_ro (H,H)
  const float* Wrs_b = (const float*)d_in[9];  const float* Wrs_w = (const float*)d_in[10];  // W_rs (H,H)
  const float* Wsr_b = (const float*)d_in[11]; const float* Wsr_w = (const float*)d_in[12];  // W_sr (D,H)
  const float* in_b  = (const float*)d_in[13]; const float* in_w  = (const float*)d_in[14];  // input_lin (6H, D+SD)
  const float* mx_b  = (const float*)d_in[15]; const float* mx_w  = (const float*)d_in[16];  // mapping_x (3H, 2H)
  const float* oo1_b = (const float*)d_in[17]; const float* oo1_w = (const float*)d_in[18];  // (512, 2H)
  const float* oo2_b = (const float*)d_in[19]; const float* oo2_w = (const float*)d_in[20];  // (1, 512)
  const float* or1_b = (const float*)d_in[21]; const float* or1_w = (const float*)d_in[22];
  const float* or2_b = (const float*)d_in[23]; const float* or2_w = (const float*)d_in[24];
  const float* pl_b  = (const float*)d_in[25]; const float* pl_w  = (const float*)d_in[26];  // post_lstm (2H, H)
  const float* rc_b  = (const float*)d_in[27]; const float* rc_w  = (const float*)d_in[28];  // rel_compress (R, 3H)
  const float* sf_b  = (const float*)d_in[29]; const float* sf_w  = (const float*)d_in[30];  // score_fc (C, H)
  const float* sr1_b = (const float*)d_in[31]; const float* sr1_w = (const float*)d_in[32];
  const float* sr2_b = (const float*)d_in[33]; const float* sr2_w = (const float*)d_in[34];
  const float* st_b  = (const float*)d_in[35]; const float* st_w  = (const float*)d_in[36];  // state_lin (5H, H)
  const float* step_w = (const float*)d_in[37];                                              // (C, SD)
  const float* pred_in = (const float*)d_in[38];   // E x H
  const int*   rel     = (const int*)d_in[39];     // E x 2

  float* ws = (float*)d_ws;
  size_t off = 0;
  auto alloc = [&](size_t n) { float* p = ws + off; off += n; return p; };
  float* soft  = alloc((size_t)Nn * Cc);
  float* stepE = alloc((size_t)Nn * SDd);
  float* state = alloc((size_t)Nn * Hh);
  float* memry = alloc((size_t)Nn * Hh);
  float* pf    = alloc((size_t)Ee * Hh);
  float* cat   = alloc((size_t)Ee * 2 * Hh);
  float* hid   = alloc((size_t)Ee * 2 * Hh);
  float* u     = alloc(Ee);
  float* uexp  = alloc(Ee);
  float* oo    = alloc((size_t)Nn * Nn);
  float* rs    = alloc(Nn);
  float* denom = alloc(Nn);
  float* ssum  = alloc(Nn);
  float* agg   = alloc((size_t)Nn * Hh);
  float* msg   = alloc((size_t)Nn * Dd);
  float* xbuf  = alloc((size_t)Nn * (Dd + SDd));
  float* pi    = alloc((size_t)Nn * 6 * Hh);
  float* ps    = alloc((size_t)Nn * 5 * Hh);
  float* srs   = alloc((size_t)Nn * Hh);
  float* sro   = alloc((size_t)Nn * Hh);
  float* ectx  = alloc((size_t)Nn * 2 * Hh);
  float* prod  = alloc((size_t)Ee * 3 * Hh);

  auto fill = [&](float* p, float v, size_t n) {
    int blocks = (int)((n + 255) / 256); if (blocks > 4096) blocks = 4096;
    fill_kernel<<<blocks, 256, 0, stream>>>(p, v, (long long)n);
  };

  // --- precompute: step_embed = softmax(obj_logits) @ step_obj_embed
  softmax_kernel<<<Nn, TPB, 0, stream>>>(obj_logit, soft, Cc);
  gemm(stream, soft, Cc, step_w, SDd, 0, nullptr, nullptr, nullptr, stepE, SDd, Nn, SDd, Cc, EPI_NONE, 0);

  fill(state, 0.0f, (size_t)Nn * Hh);
  fill(memry, 0.0f, (size_t)Nn * Hh);
  hipMemcpyAsync(pf, pred_in, (size_t)Ee * Hh * sizeof(float), hipMemcpyDeviceToDevice, stream);

  for (int it = 0; it < ITERS; ++it) {
    // ---- oo attention: u_o = MLP([state[sub], state[obj]])
    concat_gather_kernel<<<Ee, TPB, 0, stream>>>(cat, state, rel, 2, Hh, 0, Hh,
                                                 state, rel + 1, 2, Hh, 0, Hh);
    gemm(stream, cat, 2 * Hh, oo1_w, 2 * Hh, 1, oo1_b, nullptr, nullptr, hid, 512, Ee, 512, 2 * Hh, EPI_RELU, 0);
    edge_dot_kernel<<<Ee / 8, TPB, 0, stream>>>(hid, 512, oo2_w, oo2_b, u, nullptr, nullptr, 0, nullptr, Ee, 512);
    fill(oo, 0.0f, (size_t)Nn * Nn);
    scatter_oo_kernel<<<(Ee + 255) / 256, 256, 0, stream>>>(u, rel, oo, Ee, Nn);
    oo_norm_kernel<<<Nn, TPB, 0, stream>>>(oo, rs, Nn);
    // msg  = norm_oo @ (state@Wo^T + b) = (norm_oo@state)@Wo^T + rowsum(norm_oo)*b
    gemm(stream, oo, Nn, state, Hh, 0, nullptr, nullptr, nullptr, agg, Hh, Nn, Hh, Nn, EPI_NONE, 0);
    gemm(stream, agg, Hh, Wo_w, Hh, 1, Wo_b, rs, nullptr, msg, Dd, Nn, Dd, Hh, EPI_NONE, 0);

    // ---- sr attention (idx = sub)
    concat_gather_kernel<<<Ee, TPB, 0, stream>>>(cat, state, rel, 2, Hh, 0, Hh,
                                                 pf, nullptr, 0, Hh, 0, Hh);
    gemm(stream, cat, 2 * Hh, sr1_w, 2 * Hh, 1, sr1_b, nullptr, nullptr, hid, 512, Ee, 512, 2 * Hh, EPI_RELU, 0);
    fill(denom, 0.0f, Nn); fill(ssum, 0.0f, Nn); fill(agg, 0.0f, (size_t)Nn * Hh);
    edge_dot_kernel<<<Ee / 8, TPB, 0, stream>>>(hid, 512, sr2_w, sr2_b, nullptr, uexp, rel, 2, denom, Ee, 512);
    alpha_agg_kernel<<<Ee, TPB, 0, stream>>>(uexp, rel, 2, denom, pf, agg, ssum, Hh);
    gemm(stream, agg, Hh, Wsr_w, Hh, 1, Wsr_b, ssum, nullptr, msg, Dd, Nn, Dd, Hh, EPI_NONE, 1);

    // ---- or attention (idx = obj)
    concat_gather_kernel<<<Ee, TPB, 0, stream>>>(cat, state, rel + 1, 2, Hh, 0, Hh,
                                                 pf, nullptr, 0, Hh, 0, Hh);
    gemm(stream, cat, 2 * Hh, or1_w, 2 * Hh, 1, or1_b, nullptr, nullptr, hid, 512, Ee, 512, 2 * Hh, EPI_RELU, 0);
    fill(denom, 0.0f, Nn); fill(ssum, 0.0f, Nn); fill(agg, 0.0f, (size_t)Nn * Hh);
    edge_dot_kernel<<<Ee / 8, TPB, 0, stream>>>(hid, 512, or2_w, or2_b, nullptr, uexp, rel + 1, 2, denom, Ee, 512);
    alpha_agg_kernel<<<Ee, TPB, 0, stream>>>(uexp, rel + 1, 2, denom, pf, agg, ssum, Hh);
    gemm(stream, agg, Hh, Wor_w, Hh, 1, Wor_b, ssum, nullptr, msg, Dd, Nn, Dd, Hh, EPI_NONE, 1);

    // ---- LSTM-ish state update
    build_x_kernel<<<Nn, TPB, 0, stream>>>(obj_fmaps, msg, stepE, xbuf, Dd, SDd);
    gemm(stream, xbuf, Dd + SDd, in_w, Dd + SDd, 1, in_b, nullptr, nullptr, pi, 6 * Hh, Nn, 6 * Hh, Dd + SDd, EPI_NONE, 0);
    gemm(stream, state, Hh, st_w, Hh, 1, st_b, nullptr, nullptr, ps, 5 * Hh, Nn, 5 * Hh, Hh, EPI_NONE, 0);
    lstm_kernel<<<(Nn * Hh + 255) / 256, 256, 0, stream>>>(pi, ps, memry, state, Nn, Hh);

    // ---- pred_feats += state[sub]@Wrs^T + state[obj]@Wro^T (node-level GEMMs + gather-add)
    gemm(stream, state, Hh, Wrs_w, Hh, 1, Wrs_b, nullptr, nullptr, srs, Hh, Nn, Hh, Hh, EPI_NONE, 0);
    gemm(stream, state, Hh, Wro_w, Hh, 1, Wro_b, nullptr, nullptr, sro, Hh, Nn, Hh, Hh, EPI_NONE, 0);
    pf_update_kernel<<<Ee, TPB, 0, stream>>>(pf, srs, sro, rel, Hh);
  }

  // ---- finals
  float* obj_dists = (float*)d_out;                       // N x C
  float* rel_dists = (float*)d_out + (size_t)Nn * Cc;     // E x R
  gemm(stream, state, Hh, sf_w, Hh, 1, sf_b, nullptr, nullptr, obj_dists, Cc, Nn, Cc, Hh, EPI_NONE, 0);
  gemm(stream, state, Hh, pl_w, Hh, 1, pl_b, nullptr, nullptr, ectx, 2 * Hh, Nn, 2 * Hh, Hh, EPI_NONE, 0);
  concat_gather_kernel<<<Ee, TPB, 0, stream>>>(cat, ectx, rel, 2, 2 * Hh, 0, Hh,
                                               ectx, rel + 1, 2, 2 * Hh, Hh, Hh);
  gemm(stream, cat, 2 * Hh, mx_w, 2 * Hh, 1, mx_b, nullptr, last_rel, prod, 3 * Hh, Ee, 3 * Hh, 2 * Hh, EPI_MULTANH, 0);
  gemm(stream, prod, 3 * Hh, rc_w, 3 * Hh, 1, rc_b, nullptr, nullptr, rel_dists, Rr, Ee, Rr, 3 * Hh, EPI_NONE, 0);
}
